// GRUD_25958782337191
// MI455X (gfx1250) — compile-verified, matrix operand order checked by reading, not run on hardware
//
#include <hip/hip_runtime.h>
#include <math.h>

// GRU-D forward for MI455X (gfx1250, wave32, WMMA).
// B=512, T=256, D=128, H=256, C=2.
//
// Pipeline:
//  k_prep   : elementwise gamma_x / imputation; writes impute output (f32) and
//             bf16 time-major staging of inp=[x_h,m] and deltas.
//  k_pack   : pre-swizzles B-matrices ([Wih;Whh]^T and W_gh^T) into per-lane
//             WMMA bf16 fragment layout (32B contiguous per lane per fragment),
//             and fuses bih+bhh.
//  k_gammah : big parallel WMMA GEMM: gamma_h = exp(-relu(delta @ W_gh^T + b)).
//  k_recur  : persistent recurrent kernel, 32 WGs x 16 waves; per timestep one
//             fused K=512 bf16 WMMA GEMM ([inp | h*gamma_h] @ [Wih;Whh]^T),
//             LSTM pointwise in LDS, h/c state resident in LDS for all 256 steps.

#define B_   512
#define T_   256
#define D_   128
#define H_   256
#define C_   2
#define K2D  (2*D_)        // 256  (inp width)
#define KCAT (K2D + H_)    // 512  (fused GEMM K)
#define G4H  (4*H_)        // 1024 (gate width)

typedef __attribute__((ext_vector_type(16))) __bf16 v16bf;
typedef __attribute__((ext_vector_type(8)))  float  v8f;

// ---------- helpers ----------
static __device__ __forceinline__ unsigned short f2bf(float f) {
  unsigned int u = __float_as_uint(f);
  u += 0x7FFFu + ((u >> 16) & 1u);            // round-to-nearest-even
  return (unsigned short)(u >> 16);
}
static __device__ __forceinline__ float bf2f(unsigned short s) {
  return __uint_as_float(((unsigned int)s) << 16);
}
// 16-bit A-matrix 16x32 fragment layout (ISA 7.12.2): lane holds row m=lane%16;
// half=lane/16; VGPR slot v, pair p -> K = v<4 ? 8*half+2v+p : 16+8*half+2(v-4)+p.
static __device__ __forceinline__ int kloc(int lane, int v, int p) {
  int half = lane >> 4;
  return (v < 4) ? (8*half + 2*v + p) : (16 + 8*half + 2*(v-4) + p);
}
// inverse: (klocal, m) -> (lane, element slot within lane's 16 bf16)
static __device__ __forceinline__ void amap(int klocal, int m, int &lane, int &slot) {
  int half, v, p;
  if (klocal < 16) { half = klocal >> 3; int q = klocal & 7; v = q >> 1;       p = q & 1; }
  else { int kk = klocal - 16; half = kk >> 3; int q = kk & 7; v = 4 + (q >> 1); p = q & 1; }
  lane = m + (half << 4);
  slot = 2*v + p;
}
static __device__ __forceinline__ v8f wmma_bf16(v16bf a, v16bf b, v8f c) {
  return __builtin_amdgcn_wmma_f32_16x16x32_bf16(false, a, false, b, (short)0, c, false, false);
}
static __device__ __forceinline__ float sigmoidf(float x) { return 1.0f / (1.0f + expf(-x)); }

// ---------- workspace layout (bytes) ----------
static constexpr size_t OFF_INP  = 0;                                           // T*B*K2D u16
static constexpr size_t OFF_DLT  = OFF_INP + (size_t)T_*B_*K2D*2;               // T*B*D   u16
static constexpr size_t OFF_GMH  = OFF_DLT + (size_t)T_*B_*D_*2;                // T*B*H   u16
static constexpr size_t OFF_BCAT = OFF_GMH + (size_t)T_*B_*H_*2;                // 64*16*512 u16
static constexpr size_t OFF_BGH  = OFF_BCAT + (size_t)64*16*512*2;              // 16*4*512  u16
static constexpr size_t OFF_BIAS = OFF_BGH + (size_t)16*4*512*2;                // 1024 f32

// ---------------------------------------------------------------------------
// Kernel 1: elementwise imputation + bf16 staging (time-major for the scan)
// ---------------------------------------------------------------------------
__global__ void k_prep(const float* __restrict__ values, const int* __restrict__ masks,
                       const float* __restrict__ deltas, const float* __restrict__ forwards,
                       const float* __restrict__ w_gx, const float* __restrict__ b_gx,
                       float* __restrict__ impute_out,
                       unsigned short* __restrict__ inp_ws,
                       unsigned short* __restrict__ dlt_ws) {
  int n  = blockIdx.x * blockDim.x + threadIdx.x;     // over B*T*D
  int di = n % D_;
  int bt = n / D_;
  int t  = bt % T_;
  int b  = bt / T_;
  float x  = values[n];
  float m  = (float)masks[n];
  float dl = deltas[n];
  float f  = forwards[n];
  float gx = expf(-fmaxf(dl * w_gx[di] + b_gx[di], 0.0f));
  float xh = m * x + (1.0f - m) * (1.0f - gx) * f;
  impute_out[n] = xh;                                  // impute_tensor output (f32)
  size_t r = ((size_t)t * B_ + b) * K2D;
  inp_ws[r + di]       = f2bf(xh);
  inp_ws[r + D_ + di]  = f2bf(m);
  dlt_ws[((size_t)t * B_ + b) * D_ + di] = f2bf(dl);
}

// ---------------------------------------------------------------------------
// Kernel 2: pack B-matrices into per-lane WMMA fragment layout + fuse biases
// ---------------------------------------------------------------------------
__global__ void k_pack(const float* __restrict__ W_gh, const float* __restrict__ Wih,
                       const float* __restrict__ Whh, const float* __restrict__ bih,
                       const float* __restrict__ bhh,
                       unsigned short* __restrict__ bcat, unsigned short* __restrict__ bgh,
                       float* __restrict__ bias_sum) {
  int e = blockIdx.x * blockDim.x + threadIdx.x;
  const int NCAT = 64 * 16 * 512;   // 64 j-tiles x 16 k-steps x (32 lanes x 16 bf16)
  const int NGH  = 16 * 4 * 512;    // 16 n-tiles x 4 k-steps
  if (e < NCAT) {
    int frag = e >> 9, r = e & 511;
    int lane = r >> 4, sl = r & 15, v = sl >> 1, p = sl & 1;
    int jt = frag >> 4, ks = frag & 15;
    int j  = jt * 16 + (lane & 15);
    int k  = ks * 32 + kloc(lane, v, p);
    float w = (k < K2D) ? Wih[(size_t)j * K2D + k] : Whh[(size_t)j * H_ + (k - K2D)];
    bcat[e] = f2bf(w);
  } else if (e < NCAT + NGH) {
    int e2 = e - NCAT;
    int frag = e2 >> 9, r = e2 & 511;
    int lane = r >> 4, sl = r & 15, v = sl >> 1, p = sl & 1;
    int nt = frag >> 2, ks = frag & 3;
    int k  = ks * 32 + kloc(lane, v, p);
    bgh[e2] = f2bf(W_gh[(size_t)(nt * 16 + (lane & 15)) * D_ + k]);
  } else if (e < NCAT + NGH + G4H) {
    int j = e - NCAT - NGH;
    bias_sum[j] = bih[j] + bhh[j];
  }
}

// ---------------------------------------------------------------------------
// Kernel 3: gamma_h = exp(-relu(delta @ W_gh^T + b_gh)) — parallel WMMA GEMM
// [T*B, 128] x [128, 256]; 16 rows per block, 8 waves x 2 n-tiles each.
// ---------------------------------------------------------------------------
__global__ void k_gammah(const unsigned short* __restrict__ dlt,
                         const unsigned short* __restrict__ bgh_w,
                         const float* __restrict__ b_gh,
                         unsigned short* __restrict__ gmh) {
  int row0 = blockIdx.x * 16;
  int tid = threadIdx.x, wave = tid >> 5, lane = tid & 31;
  int m = lane & 15, half = lane >> 4;
  union F { v16bf v; unsigned int u[8]; } a[4];
  const unsigned int* drow = (const unsigned int*)(dlt + (size_t)(row0 + m) * D_);
#pragma unroll
  for (int ks = 0; ks < 4; ++ks)
#pragma unroll
    for (int v = 0; v < 8; ++v) {
      int kb = kloc(lane, v, 0);                       // pair base (even K)
      a[ks].u[v] = drow[(ks * 32 + kb) >> 1];          // 2 bf16 per dword
    }
  v8f acc0 = {0.f,0.f,0.f,0.f,0.f,0.f,0.f,0.f};
  v8f acc1 = {0.f,0.f,0.f,0.f,0.f,0.f,0.f,0.f};
  int nt0 = wave * 2;
#pragma unroll
  for (int ks = 0; ks < 4; ++ks) {
    v16bf b0 = *(const v16bf*)(bgh_w + (((size_t)(nt0 * 4 + ks)) << 9) + (lane << 4));
    v16bf b1 = *(const v16bf*)(bgh_w + (((size_t)((nt0 + 1) * 4 + ks)) << 9) + (lane << 4));
    acc0 = wmma_bf16(a[ks].v, b0, acc0);
    acc1 = wmma_bf16(a[ks].v, b1, acc1);
  }
  int n = lane & 15;
#pragma unroll
  for (int r = 0; r < 8; ++r) {                        // C layout: m = r + 8*half, n = lane%16
    int mm = r + 8 * half;
    size_t base = (size_t)(row0 + mm) * H_;
    gmh[base + nt0 * 16 + n]       = f2bf(expf(-fmaxf(acc0[r] + b_gh[nt0 * 16 + n], 0.f)));
    gmh[base + (nt0 + 1) * 16 + n] = f2bf(expf(-fmaxf(acc1[r] + b_gh[(nt0 + 1) * 16 + n], 0.f)));
  }
}

// ---------------------------------------------------------------------------
// Kernel 4: persistent recurrent kernel. 32 WGs x 512 threads (16 waves).
// LDS: h[16x256] f32 | c[16x256] f32 | gates[16x1024] f32 | A-frag buf 16KB.
// ---------------------------------------------------------------------------
__global__ void __launch_bounds__(512, 1)
k_recur(const unsigned short* __restrict__ inp, const unsigned short* __restrict__ gmh,
        const unsigned short* __restrict__ bcat, const float* __restrict__ bias_sum,
        const float* __restrict__ Wout, const float* __restrict__ bout,
        float* __restrict__ y_out) {
  extern __shared__ char smem[];
  float* h_s = (float*)smem;                 // 16*256
  float* c_s = h_s + 16 * H_;                // 16*256
  float* g_s = c_s + 16 * H_;                // 16*1024
  unsigned short* Ab = (unsigned short*)(g_s + 16 * G4H);  // 16 ksteps x 32 lanes x 16 bf16

  int tid = threadIdx.x, wave = tid >> 5, lane = tid & 31;
  int b0 = blockIdx.x * 16;
  for (int e = tid; e < 16 * H_; e += 512) { h_s[e] = 0.f; c_s[e] = 0.f; }
  __syncthreads();

  for (int t = 0; t < T_; ++t) {
    size_t gbase = (size_t)t * B_ + b0;
    // --- phase 1: h *= gamma_h; pack A = [inp | h] in fragment layout ---
    for (int e = tid; e < 16 * H_; e += 512) {
      int m = e >> 8, hh = e & 255;
      float hv = h_s[e] * bf2f(gmh[(gbase + m) * H_ + hh]);
      h_s[e] = hv;
      int lz, sz; amap(hh & 31, m, lz, sz);
      Ab[(((8 + (hh >> 5)) * 32 + lz) << 4) + sz] = f2bf(hv);      // K = 256+hh
    }
    for (int e = tid; e < 16 * K2D; e += 512) {
      int m = e >> 8, k = e & 255;
      unsigned short val = inp[(gbase + m) * K2D + k];
      int lz, sz; amap(k & 31, m, lz, sz);
      Ab[((((k >> 5)) * 32 + lz) << 4) + sz] = val;                // K = k
    }
    if (t + 1 < T_) {                                              // gfx1250 global_prefetch_b8
      __builtin_prefetch(&gmh[((size_t)(t + 1) * B_ + b0 + (tid & 15)) * H_], 0, 1);
      __builtin_prefetch(&inp[((size_t)(t + 1) * B_ + b0 + (tid & 15)) * K2D], 0, 1);
    }
    __syncthreads();
    // --- phase 2: gates = A[16x512] @ Bcat[512x1024]; wave -> 64 columns ---
    v8f z = {0.f,0.f,0.f,0.f,0.f,0.f,0.f,0.f};
    v8f acc[4] = {z, z, z, z};
    int jt0 = wave * 4;
    for (int ks = 0; ks < 16; ++ks) {
      v16bf a = *(const v16bf*)(Ab + ((ks * 32 + lane) << 4));     // ds_load 32B/lane
#pragma unroll
      for (int nt = 0; nt < 4; ++nt) {
        v16bf b = *(const v16bf*)(bcat + (((size_t)((jt0 + nt) * 16 + ks)) << 9) + (lane << 4));
        acc[nt] = wmma_bf16(a, b, acc[nt]);
      }
    }
#pragma unroll
    for (int nt = 0; nt < 4; ++nt)
#pragma unroll
      for (int r = 0; r < 8; ++r) {
        int mm = r + 8 * (lane >> 4);
        g_s[mm * G4H + (jt0 + nt) * 16 + (lane & 15)] = acc[nt][r];
      }
    __syncthreads();
    // --- phase 3: LSTM pointwise (gate order i,f,g,o) ---
    for (int e = tid; e < 16 * H_; e += 512) {
      int m = e >> 8, hh = e & 255;
      const float* gr = g_s + m * G4H;
      float ig = gr[hh]          + bias_sum[hh];
      float fg = gr[H_ + hh]     + bias_sum[H_ + hh];
      float gg = gr[2 * H_ + hh] + bias_sum[2 * H_ + hh];
      float og = gr[3 * H_ + hh] + bias_sum[3 * H_ + hh];
      float cv = sigmoidf(fg) * c_s[e] + sigmoidf(ig) * tanhf(gg);
      c_s[e] = cv;
      h_s[e] = sigmoidf(og) * tanhf(cv);
    }
    __syncthreads();
  }
  // --- output head: y = h @ Wout^T + bout  (C=2, tiny) ---
  if (tid < 32) {
    int m = tid >> 1, cc = tid & 1;
    float s = bout[cc];
    for (int k = 0; k < H_; ++k) s += h_s[m * H_ + k] * Wout[cc * H_ + k];
    y_out[(b0 + m) * C_ + cc] = s;
  }
}

// ---------------------------------------------------------------------------
extern "C" void kernel_launch(void* const* d_in, const int* in_sizes, int n_in,
                              void* d_out, int out_size, void* d_ws, size_t ws_size,
                              hipStream_t stream) {
  const float* values   = (const float*)d_in[0];
  const int*   masks    = (const int*)  d_in[1];
  const float* deltas   = (const float*)d_in[2];
  const float* forwards = (const float*)d_in[3];
  const float* W_gh     = (const float*)d_in[4];
  const float* b_gh     = (const float*)d_in[5];
  const float* w_gx     = (const float*)d_in[6];
  const float* b_gx     = (const float*)d_in[7];
  const float* Wih      = (const float*)d_in[8];
  const float* Whh      = (const float*)d_in[9];
  const float* bih      = (const float*)d_in[10];
  const float* bhh      = (const float*)d_in[11];
  const float* Wout     = (const float*)d_in[12];
  const float* bout     = (const float*)d_in[13];

  float* y      = (float*)d_out;                 // [B, C]
  float* impute = y + (size_t)B_ * C_;           // [B, T, D]

  char* ws = (char*)d_ws;
  unsigned short* inp_ws  = (unsigned short*)(ws + OFF_INP);
  unsigned short* dlt_ws  = (unsigned short*)(ws + OFF_DLT);
  unsigned short* gmh_ws  = (unsigned short*)(ws + OFF_GMH);
  unsigned short* bcat_ws = (unsigned short*)(ws + OFF_BCAT);
  unsigned short* bgh_ws  = (unsigned short*)(ws + OFF_BGH);
  float*          bias_ws = (float*)(ws + OFF_BIAS);

  const int ntot = B_ * T_ * D_;                 // 16,777,216
  k_prep<<<ntot / 256, 256, 0, stream>>>(values, masks, deltas, forwards, w_gx, b_gx,
                                         impute, inp_ws, dlt_ws);

  const int npack = 64 * 16 * 512 + 16 * 4 * 512 + G4H;
  k_pack<<<(npack + 255) / 256, 256, 0, stream>>>(W_gh, Wih, Whh, bih, bhh,
                                                  bcat_ws, bgh_ws, bias_ws);

  k_gammah<<<(T_ * B_) / 16, 256, 0, stream>>>(dlt_ws, bgh_ws, b_gh, gmh_ws);

  size_t smem = (size_t)(16 * H_ * 2 + 16 * G4H) * sizeof(float)
              + (size_t)16 * 32 * 16 * sizeof(unsigned short);   // 112 KB of 320 KB
  k_recur<<<B_ / 16, 512, smem, stream>>>(inp_ws, gmh_ws, bcat_ws, bias_ws,
                                          Wout, bout, y);
}
